// GCNClassifier_83588653515023
// MI455X (gfx1250) — matrix operand order repaired and checked by reading, not compiled, not run on hardware
//
#include <hip/hip_runtime.h>

#define NN 100000
#define EE 1600000
#define GG 512
#define IN_CH 96
#define HC 128
#define CC 100

typedef __attribute__((ext_vector_type(16))) __bf16 v16bf;
typedef __attribute__((ext_vector_type(8)))  __bf16 v8bf;
typedef __attribute__((ext_vector_type(8)))  float  v8f;

__device__ __forceinline__ __bf16 f2bf(float f) {
  union { float f; unsigned u; } x; x.f = f;
  unsigned r = x.u + 0x7fffu + ((x.u >> 16) & 1u);   // round-to-nearest-even
  unsigned short h = (unsigned short)(r >> 16);
  __bf16 b; __builtin_memcpy(&b, &h, 2);
  return b;
}

// ---------------- edge encoder: ew = softplus(relu(ea@W1+b1)@W2+b2)+1e-4 ----
__global__ void k_edge_weight(const float* __restrict__ ea,
                              const float* __restrict__ W1, const float* __restrict__ b1,
                              const float* __restrict__ W2, const float* __restrict__ b2,
                              float* __restrict__ ew, int E) {
  __shared__ float sW1[8 * HC];
  __shared__ float sb1[HC];
  __shared__ float sW2[HC];
  int t = threadIdx.x;
  for (int i = t; i < 8 * HC; i += blockDim.x) sW1[i] = W1[i];
  if (t < HC) { sb1[t] = b1[t]; sW2[t] = W2[t]; }
  __syncthreads();
  int e = blockIdx.x * blockDim.x + t;
  if (e >= E) return;
  float a[8];
#pragma unroll
  for (int k = 0; k < 8; ++k) a[k] = ea[e * 8 + k];
  float s = 0.f;
  for (int j = 0; j < HC; ++j) {
    float h = sb1[j];
#pragma unroll
    for (int k = 0; k < 8; ++k) h += a[k] * sW1[k * HC + j];
    h = fmaxf(h, 0.f);
    s += h * sW2[j];
  }
  s += b2[0];
  float sp = fmaxf(s, 0.f) + log1pf(expf(-fabsf(s)));   // stable softplus
  ew[e] = sp + 1e-4f;
}

// ---------------- degree / norm ---------------------------------------------
__global__ void k_deg_init(float* __restrict__ deg, int n) {
  int i = blockIdx.x * blockDim.x + threadIdx.x;
  if (i < n) deg[i] = 1.0f;               // self-loop weight
}
__global__ void k_deg_scatter(const int* __restrict__ col, const float* __restrict__ ew,
                              float* __restrict__ deg, int E) {
  int e = blockIdx.x * blockDim.x + threadIdx.x;
  if (e < E) atomicAdd(&deg[col[e]], ew[e]);
}
__global__ void k_deg_to_dis(float* __restrict__ deg, int n) {
  int i = blockIdx.x * blockDim.x + threadIdx.x;
  if (i < n) { float d = deg[i]; deg[i] = d > 0.f ? rsqrtf(d) : 0.f; }
}
__global__ void k_norm(const int* __restrict__ row, const int* __restrict__ col,
                       const float* __restrict__ ew, const float* __restrict__ dis,
                       float* __restrict__ nrm, int E) {
  int e = blockIdx.x * blockDim.x + threadIdx.x;
  if (e < E) nrm[e] = dis[row[e]] * ew[e] * dis[col[e]];
}

// ---------------- bf16 conversions ------------------------------------------
__global__ void k_f32_to_bf16(const float* __restrict__ in, __bf16* __restrict__ out, int n) {
  int i = blockIdx.x * blockDim.x + threadIdx.x;
  if (i < n) out[i] = f2bf(in[i]);
}
// W [K,Nout] f32 row-major  ->  Wt [Nout,K] bf16 (so B tiles load contiguously)
__global__ void k_wt_bf16(const float* __restrict__ W, __bf16* __restrict__ Wt, int K, int Nout) {
  int t = blockIdx.x * blockDim.x + threadIdx.x;
  if (t >= K * Nout) return;
  int k = t / Nout, n = t % Nout;
  Wt[n * K + k] = f2bf(W[t]);
}

// ---------------- dense GEMM via WMMA bf16: out[N,128] = A[N,K] @ W[K,128] --
// A row-major bf16, Wt = W^T row-major bf16 [128,K]. One wave -> 16x16 tile.
// gridDim.x = Nrows/16 (exact), blockDim = 256 (8 waves -> 128 columns).
__global__ void k_mm_wmma(const __bf16* __restrict__ A, const __bf16* __restrict__ Wt,
                          const float* __restrict__ bias, float* __restrict__ out, int K) {
  const int lane = threadIdx.x & 31;
  const int wv   = threadIdx.x >> 5;          // column tile 0..7
  const int m0   = blockIdx.x * 16;
  const int half = lane >> 4;                 // lane group 0/1
  const int mr   = lane & 15;
  const long m   = m0 + mr;                   // A row held by this lane
  const long n   = wv * 16 + mr;              // B column (= Wt row) held by this lane

  v8f c = {};
  for (int kk = 0; kk < K; kk += 32) {
    // A: lanes 0-15 K=[kk,kk+16), lanes 16-31 K=[kk+16,kk+32), contiguous per lane
    v16bf a = *(const v16bf*)(A + m * K + kk + half * 16);
    // B: lanes 0-15 K=[kk,kk+16), lanes 16-31 K=[kk+16,kk+32), contiguous per lane
    v16bf b = *(const v16bf*)(Wt + n * K + kk + half * 16);
    c = __builtin_amdgcn_wmma_f32_16x16x32_bf16(false, a, false, b, (short)0, c,
                                                false, false);
  }
  const float bn = bias ? bias[n] : 0.f;
#pragma unroll
  for (int r = 0; r < 8; ++r) {               // D: VGPR r -> row m0 + r + 8*half
    out[(long)(m0 + r + 8 * half) * HC + n] = c[r] + bn;
  }
}

// ---------------- aggregation: out = cb + dis^2*hw(self) + scatter ----------
__global__ void k_agg_init(const float* __restrict__ hw, const float* __restrict__ dis,
                           const float* __restrict__ cb, float* __restrict__ out, int n) {
  int t = blockIdx.x * blockDim.x + threadIdx.x;
  if (t >= n * HC) return;
  int i = t >> 7, j = t & 127;
  float d = dis[i];
  out[t] = cb[j] + d * d * hw[t];
}
// one wave per edge, lane handles 4 contiguous features
__global__ void k_agg_scatter(const int* __restrict__ row, const int* __restrict__ col,
                              const float* __restrict__ nrm, const float* __restrict__ hw,
                              float* __restrict__ out, int E) {
  int t = blockIdx.x * blockDim.x + threadIdx.x;
  int lane = t & 31;
  int e = t >> 5;
  if (e >= E) return;
  int r = row[e], c = col[e];
  float w = nrm[e];
  float4 v = *(const float4*)(hw + (long)r * HC + lane * 4);
  float* dst = out + (long)c * HC + lane * 4;
  atomicAdd(dst + 0, w * v.x);
  atomicAdd(dst + 1, w * v.y);
  atomicAdd(dst + 2, w * v.z);
  atomicAdd(dst + 3, w * v.w);
}

// ---------------- LayerNorm + residual + ReLU + hidden accumulation ---------
// one wave32 per node; lane owns 4 features
__global__ void k_post(const float* __restrict__ h_in, const float* __restrict__ res,
                       const float* __restrict__ gam, const float* __restrict__ bet,
                       float* __restrict__ cur_out, __bf16* __restrict__ cur_bf,
                       float* __restrict__ hsum, int n, int layer) {
  int lane = threadIdx.x & 31;
  int node = (blockIdx.x * blockDim.x + threadIdx.x) >> 5;
  if (node >= n) return;
  long base = (long)node * HC + lane * 4;
  float4 v = *(const float4*)(h_in + base);
  float s = v.x + v.y + v.z + v.w;
#pragma unroll
  for (int off = 16; off > 0; off >>= 1) s += __shfl_xor(s, off, 32);
  float mu = s * (1.0f / HC);
  float dx = v.x - mu, dy = v.y - mu, dz = v.z - mu, dw = v.w - mu;
  float q = dx * dx + dy * dy + dz * dz + dw * dw;
#pragma unroll
  for (int off = 16; off > 0; off >>= 1) q += __shfl_xor(q, off, 32);
  float rstd = rsqrtf(q * (1.0f / HC) + 1e-5f);
  float4 rv = *(const float4*)(res + base);
  int j = lane * 4;
  float o0 = fmaxf(gam[j + 0] * dx * rstd + bet[j + 0] + rv.x, 0.f);
  float o1 = fmaxf(gam[j + 1] * dy * rstd + bet[j + 1] + rv.y, 0.f);
  float o2 = fmaxf(gam[j + 2] * dz * rstd + bet[j + 2] + rv.z, 0.f);
  float o3 = fmaxf(gam[j + 3] * dw * rstd + bet[j + 3] + rv.w, 0.f);
  float4 o = make_float4(o0, o1, o2, o3);
  *(float4*)(cur_out + base) = o;
  cur_bf[base + 0] = f2bf(o0); cur_bf[base + 1] = f2bf(o1);
  cur_bf[base + 2] = f2bf(o2); cur_bf[base + 3] = f2bf(o3);
  if (layer == 0) {
    *(float4*)(hsum + base) = o;
  } else {
    float4 hs = *(const float4*)(hsum + base);
    hs.x += o0; hs.y += o1; hs.z += o2; hs.w += o3;
    *(float4*)(hsum + base) = hs;
  }
}

// ---------------- graph pooling ---------------------------------------------
__global__ void k_pool_zero(float* __restrict__ gsum, float* __restrict__ gmax,
                            float* __restrict__ cnt) {
  int t = blockIdx.x * blockDim.x + threadIdx.x;
  if (t < GG * HC) { gsum[t] = 0.f; gmax[t] = 0.f; }   // features are >=0 (post-ReLU)
  if (t < GG) cnt[t] = 0.f;
}
__global__ void k_pool_scatter(const float* __restrict__ hsum, const int* __restrict__ batch,
                               float* __restrict__ gsum, float* __restrict__ gmax,
                               float* __restrict__ cnt, int n) {
  int t = blockIdx.x * blockDim.x + threadIdx.x;
  int lane = t & 31;
  int node = t >> 5;
  if (node >= n) return;
  int b = batch[node];
  float4 v = *(const float4*)(hsum + (long)node * HC + lane * 4);
  v.x *= 0.25f; v.y *= 0.25f; v.z *= 0.25f; v.w *= 0.25f;   // mean over 4 layers
  float* gs = gsum + b * HC + lane * 4;
  atomicAdd(gs + 0, v.x); atomicAdd(gs + 1, v.y);
  atomicAdd(gs + 2, v.z); atomicAdd(gs + 3, v.w);
  unsigned* gm = (unsigned*)(gmax + b * HC + lane * 4);     // >=0 floats order as uints
  atomicMax(gm + 0, __float_as_uint(v.x)); atomicMax(gm + 1, __float_as_uint(v.y));
  atomicMax(gm + 2, __float_as_uint(v.z)); atomicMax(gm + 3, __float_as_uint(v.w));
  if (lane == 0) atomicAdd(cnt + b, 1.0f);
}

// ---------------- head: relu([mean,max]@hW1+b1)@hW2+b2 ----------------------
__global__ void k_head(const float* __restrict__ gsum, const float* __restrict__ gmax,
                       const float* __restrict__ cnt,
                       const float* __restrict__ hW1, const float* __restrict__ hb1,
                       const float* __restrict__ hW2, const float* __restrict__ hb2,
                       float* __restrict__ out) {
  __shared__ float gv[2 * HC];
  __shared__ float h1[HC];
  int g = blockIdx.x, t = threadIdx.x;     // blockDim = 128
  float c = fmaxf(cnt[g], 1.0f);
  gv[t] = gsum[g * HC + t] / c;
  gv[HC + t] = gmax[g * HC + t];
  __syncthreads();
  float acc = hb1[t];
  for (int k = 0; k < 2 * HC; ++k) acc += gv[k] * hW1[k * HC + t];
  h1[t] = fmaxf(acc, 0.f);
  __syncthreads();
  if (t < CC) {
    float o = hb2[t];
    for (int k = 0; k < HC; ++k) o += h1[k] * hW2[k * CC + t];
    out[g * CC + t] = o;
  }
}

extern "C" void kernel_launch(void* const* d_in, const int* in_sizes, int n_in,
                              void* d_out, int out_size, void* d_ws, size_t ws_size,
                              hipStream_t stream) {
  const float* x     = (const float*)d_in[0];
  const int*   ei    = (const int*)d_in[1];
  const int*   batch = (const int*)d_in[2];
  const float* eattr = (const float*)d_in[3];
  const float* eeW1  = (const float*)d_in[4];
  const float* eeb1  = (const float*)d_in[5];
  const float* eeW2  = (const float*)d_in[6];
  const float* eeb2  = (const float*)d_in[7];
  const float* W[4]  = {(const float*)d_in[8],  (const float*)d_in[10],
                        (const float*)d_in[12], (const float*)d_in[14]};
  const float* cb[4] = {(const float*)d_in[9],  (const float*)d_in[11],
                        (const float*)d_in[13], (const float*)d_in[15]};
  const float* gam[4] = {(const float*)d_in[16], (const float*)d_in[18],
                         (const float*)d_in[20], (const float*)d_in[22]};
  const float* bet[4] = {(const float*)d_in[17], (const float*)d_in[19],
                         (const float*)d_in[21], (const float*)d_in[23]};
  const float* resW = (const float*)d_in[24];
  const float* resb = (const float*)d_in[25];
  const float* hW1  = (const float*)d_in[26];
  const float* hb1  = (const float*)d_in[27];
  const float* hW2  = (const float*)d_in[28];
  const float* hb2  = (const float*)d_in[29];
  float* out = (float*)d_out;
  (void)in_sizes; (void)n_in; (void)out_size; (void)ws_size;

  const int* row = ei;
  const int* col = ei + EE;

  // workspace carve-out (256B aligned)
  char* p = (char*)d_ws;
  size_t off = 0;
  auto take = [&](size_t bytes) -> void* {
    void* r = p + off;
    off = (off + bytes + 255) & ~(size_t)255;
    return r;
  };
  float*  ew   = (float*)take((size_t)EE * 4);
  float*  deg  = (float*)take((size_t)NN * 4);          // becomes dis in place
  float*  nrm  = (float*)take((size_t)EE * 4);
  __bf16* xbf  = (__bf16*)take((size_t)NN * IN_CH * 2);
  __bf16* wt0  = (__bf16*)take((size_t)IN_CH * HC * 2);
  __bf16* wt1  = (__bf16*)take((size_t)HC * HC * 2);
  __bf16* wt2  = (__bf16*)take((size_t)HC * HC * 2);
  __bf16* wt3  = (__bf16*)take((size_t)HC * HC * 2);
  __bf16* wtr  = (__bf16*)take((size_t)IN_CH * HC * 2);
  float*  hw   = (float*)take((size_t)NN * HC * 4);
  float*  agg  = (float*)take((size_t)NN * HC * 4);
  float*  cur0 = (float*)take((size_t)NN * HC * 4);
  float*  cur1 = (float*)take((size_t)NN * HC * 4);     // holds res-projection first
  float*  hsum = (float*)take((size_t)NN * HC * 4);
  __bf16* cbf  = (__bf16*)take((size_t)NN * HC * 2);
  float*  gsum = (float*)take((size_t)GG * HC * 4);
  float*  gmax = (float*)take((size_t)GG * HC * 4);
  float*  cnt  = (float*)take((size_t)GG * 4);
  __bf16* wt[4] = {wt0, wt1, wt2, wt3};

  // edge weights, degrees, symmetric norm
  k_edge_weight<<<EE / 256, 256, 0, stream>>>(eattr, eeW1, eeb1, eeW2, eeb2, ew, EE);
  k_deg_init<<<(NN + 255) / 256, 256, 0, stream>>>(deg, NN);
  k_deg_scatter<<<EE / 256, 256, 0, stream>>>(col, ew, deg, EE);
  k_deg_to_dis<<<(NN + 255) / 256, 256, 0, stream>>>(deg, NN);
  k_norm<<<EE / 256, 256, 0, stream>>>(row, col, ew, deg, nrm, EE);

  // bf16 conversions (x, transposed conv weights)
  k_f32_to_bf16<<<(NN * IN_CH + 255) / 256, 256, 0, stream>>>(x, xbf, NN * IN_CH);
  k_wt_bf16<<<(IN_CH * HC + 255) / 256, 256, 0, stream>>>(W[0], wt0, IN_CH, HC);
  k_wt_bf16<<<(HC * HC + 255) / 256, 256, 0, stream>>>(W[1], wt1, HC, HC);
  k_wt_bf16<<<(HC * HC + 255) / 256, 256, 0, stream>>>(W[2], wt2, HC, HC);
  k_wt_bf16<<<(HC * HC + 255) / 256, 256, 0, stream>>>(W[3], wt3, HC, HC);
  k_wt_bf16<<<(IN_CH * HC + 255) / 256, 256, 0, stream>>>(resW, wtr, IN_CH, HC);

  // residual projection for layer 0:  cur1 = x @ resW + resb   (WMMA)
  k_mm_wmma<<<NN / 16, 256, 0, stream>>>(xbf, wtr, resb, cur1, IN_CH);

  // 4 GCN layers
  for (int l = 0; l < 4; ++l) {
    const __bf16* Abf = (l == 0) ? xbf : cbf;
    const int K = (l == 0) ? IN_CH : HC;
    k_mm_wmma<<<NN / 16, 256, 0, stream>>>(Abf, wt[l], nullptr, hw, K);     // hw = h@W
    k_agg_init<<<(NN * HC + 255) / 256, 256, 0, stream>>>(hw, deg, cb[l], agg, NN);
    k_agg_scatter<<<(EE * 32) / 256, 256, 0, stream>>>(row, col, nrm, hw, agg, EE);
    float* res = (l % 2 == 0) ? cur1 : cur0;
    float* nxt = (l % 2 == 0) ? cur0 : cur1;
    k_post<<<(NN * 32 + 255) / 256, 256, 0, stream>>>(agg, res, gam[l], bet[l],
                                                      nxt, cbf, hsum, NN, l);
  }

  // pooling + head
  k_pool_zero<<<(GG * HC + 255) / 256, 256, 0, stream>>>(gsum, gmax, cnt);
  k_pool_scatter<<<(NN * 32 + 255) / 256, 256, 0, stream>>>(hsum, batch, gsum, gmax, cnt, NN);
  k_head<<<GG, HC, 0, stream>>>(gsum, gmax, cnt, hW1, hb1, hW2, hb2, out);
}
